// SelectiveSSM_13563506721344
// MI455X (gfx1250) — matrix operand order repaired and checked by reading, not compiled, not run on hardware
//
#include <hip/hip_runtime.h>
#include <hip/hip_bf16.h>

// ---------------------------------------------------------------------------
// Selective SSM (Mamba-style) forward for MI455X / gfx1250.
//   - All GEMMs: v_wmma_f32_16x16x32_bf16, f32 accumulate.
//   - bf16 operands pre-converted ONCE (weights/activations) so the GEMM tile
//     path is pure data movement (no VALU convert competing with WMMA).
//   - Tile staging via Tensor Data Mover (tensor_load_to_lds + TENSORcnt),
//     with TDM padding giving an 80B LDS row stride (bank-conflict free).
//   - Sequential scan: state in VGPRs, B/C broadcast via LDS, emits bf16 y.
// ---------------------------------------------------------------------------

#define BM 128
#define BN 128
#define BK 32
#define LDSTR 40   // bf16 elems per LDS row: 80B = 20 banks, conflict-free

typedef __attribute__((ext_vector_type(16))) __bf16 v16bf;
typedef __attribute__((ext_vector_type(8)))  float  v8f;
typedef __attribute__((ext_vector_type(4)))  unsigned int u32x4;
typedef __attribute__((ext_vector_type(8)))  int          i32x8;
typedef __attribute__((ext_vector_type(4)))  int          i32x4;

union Frag16 { unsigned int u[8]; v16bf v; };

__device__ __forceinline__ unsigned short f2bf(float f) {
  unsigned int a = __float_as_uint(f);
  return (unsigned short)((a + 0x7FFFu + ((a >> 16) & 1u)) >> 16);
}
__device__ __forceinline__ unsigned int pk_bf16(float lo, float hi) {
  return (unsigned int)f2bf(lo) | ((unsigned int)f2bf(hi) << 16);
}

#if __has_builtin(__builtin_amdgcn_tensor_load_to_lds)
#define HAVE_TDM 1
// Issue a 2-D TDM tile load: tile_d0 x tile_d1 elements of 2 bytes from a
// row-major [rows][rowlen] bf16 tensor into LDS at lds_addr, padding each
// 64B LDS row with 16B (-> LDSTR=40 bf16 stride). OOB rows read as zero.
// This toolchain's builtin is the 6-arg form:
//   (u32x4 g0, i32x8 g1, i32x4 g2, i32x4 g3, i32x8 g4, i32 cpol)
__device__ __forceinline__ void tdm_load_tile_bf16(
    unsigned int lds_addr, unsigned long long gaddr,
    unsigned int rowlen, unsigned int rows,
    unsigned int tile_d0, unsigned int tile_d1)
{
  u32x4 g0;
  g0[0] = 1u;                                       // count=1, user descriptor
  g0[1] = lds_addr;                                 // LDS byte address
  g0[2] = (unsigned int)(gaddr & 0xFFFFFFFFull);    // global_addr[31:0]
  g0[3] = (unsigned int)((gaddr >> 32) & 0x1FFFFFFull) | (2u << 30); // type=2

  i32x8 g1;
  // data_size=1 (2B) | pad_enable | pad_interval=3 (16 DW) | pad_amount=3 (4 DW)
  g1[0] = (int)((1u << 16) | (1u << 20) | (3u << 22) | (3u << 25));
  g1[1] = (int)((rowlen & 0xFFFFu) << 16);          // tensor_dim0[15:0]
  g1[2] = (int)((rowlen >> 16) | ((rows & 0xFFFFu) << 16)); // td0 hi | td1 lo
  g1[3] = (int)((rows >> 16) | (tile_d0 << 16));    // td1 hi | tile_dim0
  g1[4] = (int)(tile_d1 & 0xFFFFu);                 // tile_dim1 (tile_dim2=0)
  g1[5] = (int)rowlen;                              // tensor_dim0_stride[31:0]
  g1[6] = 0;                                        // stride0 hi | stride1 lo
  g1[7] = 0;                                        // stride1 hi
  i32x4 gz4 = {0, 0, 0, 0};
  i32x8 gz8 = {0, 0, 0, 0, 0, 0, 0, 0};
  __builtin_amdgcn_tensor_load_to_lds(g0, g1, gz4, gz4, gz8, 0);
}
#else
#define HAVE_TDM 0
#endif

// ---------------------------------------------------------------------------
// f32 -> bf16 conversion (one-shot, bandwidth-bound).
// dst is compact row-major [rows][cols]; src rows have stride sstride floats.
// ---------------------------------------------------------------------------
__global__ __launch_bounds__(256) void conv_bf16_kernel(
    const float* __restrict__ src, unsigned short* __restrict__ dst,
    long sstride, int cols, int nquads)
{
  const int idx = blockIdx.x * 256 + threadIdx.x;
  if (idx >= nquads) return;
  const int qpr = cols >> 2;
  const int row = idx / qpr;
  const int c   = (idx - row * qpr) << 2;
  const float4 v = *(const float4*)(src + (size_t)row * sstride + c);
  unsigned int* d = (unsigned int*)(dst + (size_t)row * cols + c);
  d[0] = pk_bf16(v.x, v.y);
  d[1] = pk_bf16(v.z, v.w);
}

// ---------------------------------------------------------------------------
// C = act(A[M x K] * W[Ntot x K]^T + bias), tiled bf16 WMMA GEMM.
// A and W are compact bf16 (row length == K).  epi: 0=identity, 1=softplus.
// ---------------------------------------------------------------------------
__global__ __launch_bounds__(256) void wmma_gemm_bias(
    const unsigned short* __restrict__ A,
    const unsigned short* __restrict__ W,
    const float* __restrict__ bias,
    float* __restrict__ C, long ldc,
    int K, int Ntot, int epi)
{
  __shared__ unsigned short sA[BM * LDSTR];
  __shared__ unsigned short sB[BN * LDSTR];

  const int tid  = threadIdx.x;
  const int lane = tid & 31;
  const int l15  = lane & 15;
  const int lh   = lane >> 4;           // half-wave select
  const int w    = tid >> 5;            // wave id 0..7
  const int wm   = w >> 2;              // 2 waves along M (64 rows each)
  const int wn   = w & 3;               // 4 waves along N (32 cols each)

  const int m0 = blockIdx.y * BM;
  const int n0 = blockIdx.x * BN;

  const v8f zacc = {0.f,0.f,0.f,0.f,0.f,0.f,0.f,0.f};
  v8f acc[4][2];
  #pragma unroll
  for (int mt = 0; mt < 4; ++mt)
    #pragma unroll
    for (int nt = 0; nt < 2; ++nt) acc[mt][nt] = zacc;

#if HAVE_TDM
  const unsigned int ldsA = (unsigned int)(unsigned long long)(&sA[0]);
  const unsigned int ldsB = (unsigned int)(unsigned long long)(&sB[0]);
#else
  const int trow = tid >> 1;            // 0..127: 2 threads per tile row
  const int thal = tid & 1;             // 16-elem (32B) half select
#endif

  for (int k0 = 0; k0 < K; k0 += BK) {
    __syncthreads();   // previous iteration's LDS reads complete
#if HAVE_TDM
    if (w == 0) {      // one wave drives the Tensor Data Mover for the block
      tdm_load_tile_bf16(ldsA,
          (unsigned long long)(const void*)A + ((size_t)m0 * K + k0) * 2ull,
          (unsigned int)K, 4096u, BK, BM);
      tdm_load_tile_bf16(ldsB,
          (unsigned long long)(const void*)W + ((size_t)n0 * K + k0) * 2ull,
          (unsigned int)K, (unsigned int)Ntot, BK, BN);
      __builtin_amdgcn_s_wait_tensorcnt(0);
    }
#else
    {
      const uint4* ag = (const uint4*)(A + (size_t)(m0 + trow) * K + k0 + thal * 16);
      uint4 a0 = ag[0], a1 = ag[1];
      uint4 b0 = {0,0,0,0}, b1 = {0,0,0,0};
      const int nrow = n0 + trow;
      if (nrow < Ntot) {
        const uint4* bg = (const uint4*)(W + (size_t)nrow * K + k0 + thal * 16);
        b0 = bg[0]; b1 = bg[1];
      }
      if (k0 + BK < K) {
        __builtin_prefetch(A + (size_t)(m0 + trow) * K + (k0 + BK) + thal * 16, 0, 3);
        if (nrow < Ntot)
          __builtin_prefetch(W + (size_t)nrow * K + (k0 + BK) + thal * 16, 0, 3);
      }
      uint4* da = (uint4*)&sA[trow * LDSTR + thal * 16];
      da[0] = a0; da[1] = a1;
      uint4* db = (uint4*)&sB[trow * LDSTR + thal * 16];
      db[0] = b0; db[1] = b1;
    }
#endif
    __syncthreads();   // tile visible to all waves

    // ---- fragments per ISA 16-bit A(16x32) / B(32x16) lane layouts ----
    Frag16 fa[4];
    #pragma unroll
    for (int mt = 0; mt < 4; ++mt) {
      const int row = wm * 64 + mt * 16 + l15;
      const unsigned int* base = (const unsigned int*)&sA[row * LDSTR];
      #pragma unroll
      for (int p = 0; p < 8; ++p) {
        const int kk = (p < 4) ? (2 * p + 8 * lh) : (16 + 2 * (p - 4) + 8 * lh);
        fa[mt].u[p] = base[kk >> 1];
      }
    }
    Frag16 fb[2];
    #pragma unroll
    for (int nt = 0; nt < 2; ++nt) {
      const int col = wn * 32 + nt * 16 + l15;
      const unsigned int* base = (const unsigned int*)&sB[col * LDSTR];
      #pragma unroll
      for (int p = 0; p < 8; ++p) {
        const int kk = (p < 4) ? (2 * p + 8 * lh) : (16 + 2 * (p - 4) + 8 * lh);
        fb[nt].u[p] = base[kk >> 1];
      }
    }

    #pragma unroll
    for (int mt = 0; mt < 4; ++mt)
      #pragma unroll
      for (int nt = 0; nt < 2; ++nt)
        acc[mt][nt] = __builtin_amdgcn_wmma_f32_16x16x32_bf16(
            false, fa[mt].v, false, fb[nt].v,
            (short)0, acc[mt][nt], false, false);
  }

  // ---- epilogue: bias (+ optional softplus), guarded by Ntot ----
  #pragma unroll
  for (int mt = 0; mt < 4; ++mt) {
    #pragma unroll
    for (int nt = 0; nt < 2; ++nt) {
      const int col = n0 + wn * 32 + nt * 16 + l15;
      if (col < Ntot) {
        const float bv = bias[col];
        #pragma unroll
        for (int i = 0; i < 8; ++i) {
          const int row = m0 + wm * 64 + mt * 16 + 8 * lh + i;
          float v = acc[mt][nt][i] + bv;
          if (epi == 1) v = (v > 20.f) ? v : log1pf(__expf(v));
          C[(size_t)row * ldc + col] = v;
        }
      }
    }
  }
}

// ---------------------------------------------------------------------------
// Sequential selective scan: one thread per (batch, channel), 16-state
// recurrence in VGPRs, per-step B/C (32 floats) broadcast through LDS.
// Emits gated y directly in bf16 (feeds the output WMMA GEMM).
// ---------------------------------------------------------------------------
__global__ __launch_bounds__(256) void ssm_scan_kernel(
    const float* __restrict__ xz,     // [4096][4096]  (x | z)
    const float* __restrict__ dbc,    // [4096][96]    (delta_raw | B | C)
    const float* __restrict__ delta,  // [4096][2048]
    unsigned short* __restrict__ yb,  // out: y bf16 [4096][2048]
    const float* __restrict__ A_log,  // [2048][16]
    const float* __restrict__ Dvec)   // [2048]
{
  const int T = 2048;
  const int tid  = threadIdx.x;
  const int chan = blockIdx.x * 256 + tid;   // 0..4095
  const int b    = chan >> 11;               // batch
  const int d    = chan & 2047;              // inner channel

  __shared__ float sBC[32];

  float Arow[16], h[16];
  #pragma unroll
  for (int n = 0; n < 16; ++n) {
    Arow[n] = -__expf(A_log[d * 16 + n]);
    h[n] = 0.f;
  }
  const float Dd = Dvec[d];

  for (int t = 0; t < T; ++t) {
    const int row = b * T + t;
    if (tid < 32) sBC[tid] = dbc[(size_t)row * 96 + 64 + tid];
    __syncthreads();

    const float dt = delta[(size_t)row * 2048 + d];
    const float xt = xz[(size_t)row * 4096 + d];
    const float zt = xz[(size_t)row * 4096 + 2048 + d];

    float y = 0.f;
    #pragma unroll
    for (int n = 0; n < 16; ++n) {
      const float dA = __expf(dt * Arow[n]);
      h[n] = h[n] * dA + xt * (dt * sBC[n]);
      y += h[n] * sBC[16 + n];
    }
    y += xt * Dd;
    y *= zt * (1.0f / (1.0f + __expf(-zt)));   // y * silu(z)
    yb[(size_t)row * 2048 + d] = f2bf(y);
    __syncthreads();
  }
}

// ---------------------------------------------------------------------------
extern "C" void kernel_launch(void* const* d_in, const int* in_sizes, int n_in,
                              void* d_out, int out_size, void* d_ws, size_t ws_size,
                              hipStream_t stream) {
  (void)in_sizes; (void)n_in; (void)out_size; (void)ws_size;

  const float* u     = (const float*)d_in[0];   // [2][2048][1024]
  const float* W_in  = (const float*)d_in[1];   // [4096][1024]
  const float* b_in  = (const float*)d_in[2];   // [4096]
  const float* W_x   = (const float*)d_in[3];   // [96][2048]
  const float* b_x   = (const float*)d_in[4];   // [96]
  const float* W_dt  = (const float*)d_in[5];   // [2048][64]
  const float* b_dt  = (const float*)d_in[6];   // [2048]
  const float* A_log = (const float*)d_in[7];   // [2048][16]
  const float* Dvec  = (const float*)d_in[8];   // [2048]
  const float* W_out = (const float*)d_in[9];   // [1024][2048]
  const float* b_out = (const float*)d_in[10];  // [1024]
  float* out = (float*)d_out;                   // [2][2048][1024]

  // ---- workspace carve-out (256B aligned blocks) ----
  char* ws = (char*)d_ws;
  size_t off = 0;
  auto carve = [&](size_t bytes) {
    void* p = ws + off;
    off += (bytes + 255) & ~(size_t)255;
    return p;
  };
  float*          xz     = (float*)         carve((size_t)4096*4096*4); // 64 MB
  float*          dbc    = (float*)         carve((size_t)4096*96*4);
  float*          dly    = (float*)         carve((size_t)4096*2048*4); // 32 MB
  unsigned short* u_bf   = (unsigned short*)carve((size_t)4096*1024*2);
  unsigned short* Win_bf = (unsigned short*)carve((size_t)4096*1024*2);
  unsigned short* Wx_bf  = (unsigned short*)carve((size_t)96*2048*2);
  unsigned short* Wdt_bf = (unsigned short*)carve((size_t)2048*64*2);
  unsigned short* Wout_bf= (unsigned short*)carve((size_t)1024*2048*2);
  unsigned short* x_bf   = (unsigned short*)carve((size_t)4096*2048*2); // 16 MB
  unsigned short* dr_bf  = (unsigned short*)carve((size_t)4096*64*2);
  unsigned short* y_bf   = (unsigned short*)carve((size_t)4096*2048*2); // 16 MB

  dim3 blk(256);
  auto cgrid = [](long n) { return dim3((unsigned)((n + 255) / 256)); };

  // ---- one-shot bf16 conversions of inputs ----
  conv_bf16_kernel<<<cgrid(4096L*1024/4), blk, 0, stream>>>(u,     u_bf,   1024, 1024, 4096*1024/4);
  conv_bf16_kernel<<<cgrid(4096L*1024/4), blk, 0, stream>>>(W_in,  Win_bf, 1024, 1024, 4096*1024/4);
  conv_bf16_kernel<<<cgrid(96L*2048/4),   blk, 0, stream>>>(W_x,   Wx_bf,  2048, 2048, 96*2048/4);
  conv_bf16_kernel<<<cgrid(2048L*64/4),   blk, 0, stream>>>(W_dt,  Wdt_bf, 64,   64,   2048*64/4);
  conv_bf16_kernel<<<cgrid(1024L*2048/4), blk, 0, stream>>>(W_out, Wout_bf,2048, 2048, 1024*2048/4);

  // 1) xz = u @ W_in^T + b_in          (M=4096, N=4096, K=1024)
  wmma_gemm_bias<<<dim3(32, 32), blk, 0, stream>>>(
      u_bf, Win_bf, b_in, xz, 4096, 1024, 4096, 0);

  //    x (first half of xz) -> bf16
  conv_bf16_kernel<<<cgrid(4096L*2048/4), blk, 0, stream>>>(xz, x_bf, 4096, 2048, 4096*2048/4);

  // 2) dbc = x @ W_x^T + b_x           (M=4096, N=96,   K=2048)
  wmma_gemm_bias<<<dim3(1, 32), blk, 0, stream>>>(
      x_bf, Wx_bf, b_x, dbc, 96, 2048, 96, 0);

  //    delta_raw (first 64 cols of dbc) -> bf16
  conv_bf16_kernel<<<cgrid(4096L*64/4), blk, 0, stream>>>(dbc, dr_bf, 96, 64, 4096*64/4);

  // 3) delta = softplus(delta_raw @ W_dt^T + b_dt)   (M=4096, N=2048, K=64)
  wmma_gemm_bias<<<dim3(16, 32), blk, 0, stream>>>(
      dr_bf, Wdt_bf, b_dt, dly, 2048, 64, 2048, 1);

  // 4) selective scan + skip + SiLU gating -> y (bf16)
  ssm_scan_kernel<<<dim3(16), blk, 0, stream>>>(xz, dbc, dly, y_bf, A_log, Dvec);

  // 5) out = y @ W_out^T + b_out       (M=4096, N=1024, K=2048)
  wmma_gemm_bias<<<dim3(8, 32), blk, 0, stream>>>(
      y_bf, Wout_bf, b_out, out, 1024, 2048, 1024, 0);
}
